// TwoStream_63660005261867
// MI455X (gfx1250) — matrix operand order repaired and checked by reading, not compiled, not run on hardware
//
#include <hip/hip_runtime.h>
#include <cstdint>
#include <cstddef>

// Problem constants (match reference)
#define BQ  4
#define CCH 512
#define ICH 64
#define NSP 4096   // H*W

typedef __bf16 bf16_t;
typedef __attribute__((ext_vector_type(16))) __bf16 v16bf;
typedef __attribute__((ext_vector_type(8)))  __bf16 v8bf;
typedef __attribute__((ext_vector_type(8)))  float  v8f;

__device__ __forceinline__ float bf2f(bf16_t v) {
  unsigned short u = __builtin_bit_cast(unsigned short, v);
  unsigned int w = ((unsigned int)u) << 16;
  return __builtin_bit_cast(float, w);
}
__device__ __forceinline__ bf16_t f2bf(float f) {
  unsigned int u = __builtin_bit_cast(unsigned int, f);
  unsigned int r = u + 0x7FFFu + ((u >> 16) & 1u);   // round-to-nearest-even
  return __builtin_bit_cast(bf16_t, (unsigned short)(r >> 16));
}

// ---------------------------------------------------------------------------
// Tensor Data Mover: DMA one contiguous row (1D tile) from global into LDS.
// D# built per CDNA5 ISA ch.8: group0 = {count=1 | lds_addr | global_addr | type=2},
// group1 = {mask=0, data_size, tensor_dim0, tile_dim0, dim0_stride}. Groups 2/3
// zero (<=2D tensor). Tracked by TENSORcnt.
// This toolchain exposes the 6-arg builtin:
//   (uint32x4 g0, int32x8 g1, int32x4, int32x4, int32x8, i32 cpol)
// elem_code: 1 = 2-byte elements, 2 = 4-byte elements. nelem < 65536.
// ---------------------------------------------------------------------------
#if __has_builtin(__builtin_amdgcn_tensor_load_to_lds)
#define HAVE_TDM 1
typedef __attribute__((ext_vector_type(4))) unsigned int u32x4_t;
typedef __attribute__((ext_vector_type(8))) int i32x8_t;
typedef __attribute__((ext_vector_type(4))) int i32x4_t;

__device__ __forceinline__ void tdm_load_row_to_lds(unsigned lds_addr, const void* gptr,
                                                    unsigned nelem, unsigned elem_code) {
  unsigned long long ga = (unsigned long long)(uintptr_t)gptr;
  u32x4_t g0;
  g0[0] = 1u;                                        // count=1, load, user mode
  g0[1] = lds_addr;                                  // LDS byte address
  g0[2] = (unsigned)(ga & 0xFFFFFFFFu);              // global_addr[31:0]
  g0[3] = (unsigned)((ga >> 32) & 0x1FFFFFFu)        // global_addr[56:32]
        | (2u << 30);                                // type=2 ("image")
  i32x8_t g1;
  g1[0] = (int)(elem_code << 16);                    // workgroup_mask=0, data_size
  g1[1] = (int)(nelem << 16);                        // tensor_dim0[15:0] (<<16)
  g1[2] = (int)(1u << 16);                           // tensor_dim0 hi=0 | tensor_dim1=1
  g1[3] = (int)(nelem << 16);                        // tensor_dim1 hi=0 | tile_dim0
  g1[4] = 1;                                         // tile_dim1=1, tile_dim2=0
  g1[5] = (int)nelem;                                // tensor_dim0_stride[31:0]
  g1[6] = 0;
  g1[7] = 0;
  i32x4_t z4 = {0, 0, 0, 0};
  i32x8_t z8 = {0, 0, 0, 0, 0, 0, 0, 0};
  __builtin_amdgcn_tensor_load_to_lds(g0, g1, z4, z4, z8, 0);
}
#else
#define HAVE_TDM 0
#endif

// ---------------------------------------------------------------------------
// Elementwise converts
// ---------------------------------------------------------------------------
__global__ void k_f32_to_bf16(const float* __restrict__ in, bf16_t* __restrict__ out, int n) {
  for (int i = blockIdx.x * blockDim.x + threadIdx.x; i < n; i += gridDim.x * blockDim.x)
    out[i] = f2bf(in[i]);
}

// in [rows][cols] f32 -> out [cols][rows] bf16, LDS-tiled for coalesced IO
__global__ __launch_bounds__(256) void k_transpose_f32_to_bf16(
    const float* __restrict__ in, bf16_t* __restrict__ out, int rows, int cols) {
  __shared__ float tile[32][33];
  const int c0 = blockIdx.x * 32;
  const int r0 = blockIdx.y * 32;
  const int tx = threadIdx.x;        // 0..31
  const int ty = threadIdx.y;        // 0..7
#pragma unroll
  for (int yy = ty; yy < 32; yy += 8) {
    int r = r0 + yy, c = c0 + tx;
    tile[yy][tx] = (r < rows && c < cols) ? in[(size_t)r * cols + c] : 0.0f;
  }
  __syncthreads();
#pragma unroll
  for (int yy = ty; yy < 32; yy += 8) {
    int c = c0 + yy, r = r0 + tx;
    if (c < cols && r < rows) out[(size_t)c * rows + r] = f2bf(tile[tx][yy]);
  }
}

// ---------------------------------------------------------------------------
// NT WMMA bf16 GEMM: Out = A[M,K] @ B[K,N], with B supplied TRANSPOSED:
//   Bt[n][k] row-major (ldbt = leading dim in k).
// Both A and Bt fragments are contiguous 32-byte per-lane loads.
// Wave tile 64x32 (4x2 tiles of 16x16), 8 waves/block arranged 2(M) x 4(N)
// => block tile 128x128.  K % 32 == 0; M % 64 == 0; N % 32 == 0 (all true here),
// so tiles are never partial; wave-uniform early exit keeps EXEC all-ones.
//
// OUT_MODE: 0 bf16 normal      Out[m*ldc+n]        (LDS-transposed, coalesced)
//           1 bf16 transposed  Out[n*ldc+m]        (v8bf packed stores)
//           2 f32  normal      Out[m*ldc+n]        (LDS-transposed, v8f stores)
//           3 f32  transposed fused: Out[n*ldc+m] = (*scale)*val + 2*xadd[...]
//           4 f32  normal accumulate: Out[m*ldc+n] += (*scale)*val (LDS-transposed RMW)
// BIAS_MODE: 0 none, 1 per-row bias[m], 2 per-col bias[n]
// ---------------------------------------------------------------------------
template<int OUT_MODE, int BIAS_MODE>
__global__ __launch_bounds__(256) void gemm_bf16_wmma(
    const bf16_t* __restrict__ A, const bf16_t* __restrict__ Bt,
    void* __restrict__ Out, const float* __restrict__ bias,
    const float* __restrict__ scale, const float* __restrict__ xadd,
    int M, int N, int K, int lda, int ldbt, int ldc)
{
  const int lane = threadIdx.x & 31;
  const int wave = threadIdx.x >> 5;
  const int wm = wave >> 2;          // 0..1
  const int wn = wave & 3;           // 0..3
  const int m_wave = blockIdx.y * 128 + wm * 64;
  const int n_wave = blockIdx.x * 128 + wn * 32;
  if (m_wave >= M || n_wave >= N) return;   // wave-uniform (EXEC all-1 at WMMA)

  const int r  = lane & 15;
  const int hi = lane >> 4;

  v8f acc[4][2];
#pragma unroll
  for (int i = 0; i < 4; i++)
#pragma unroll
    for (int j = 0; j < 2; j++)
#pragma unroll
      for (int e = 0; e < 8; e++) acc[i][j][e] = 0.0f;

  int arow[4];
#pragma unroll
  for (int i = 0; i < 4; i++) { int m = m_wave + i * 16 + r; arow[i] = (m < M) ? m : (M - 1); }
  int brow[2];
#pragma unroll
  for (int j = 0; j < 2; j++) { int n = n_wave + j * 16 + r; brow[j] = (n < N) ? n : (N - 1); }

  for (int k0 = 0; k0 < K; k0 += 32) {
    if (k0 + 64 < K) {   // cover latency of the streaming operands (global_prefetch_b8)
      __builtin_prefetch(A  + (size_t)arow[0] * lda  + k0 + 64 + hi * 8,  0, 1);
      __builtin_prefetch(Bt + (size_t)brow[0] * ldbt + k0 + 64 + hi * 16, 0, 1);
    }
    // A 16x32 fragment: lane holds K in [k0+hi*16, +16)
    v16bf afr[4];
#pragma unroll
    for (int i = 0; i < 4; i++) {
      const bf16_t* ap = A + (size_t)arow[i] * lda + k0 + hi * 16;
      v8bf lo = *(const v8bf*)(ap);
      v8bf hh = *(const v8bf*)(ap + 8);
      v16bf av;
#pragma unroll
      for (int e = 0; e < 8; e++) { av[e] = lo[e]; av[8 + e] = hh[e]; }
      afr[i] = av;
    }
    // B 32x16 fragment from Bt: lane = column n, K sequential -> 16 contiguous bf16
    v16bf bfr[2];
#pragma unroll
    for (int j = 0; j < 2; j++)
      bfr[j] = *(const v16bf*)(Bt + (size_t)brow[j] * ldbt + k0 + hi * 16);

#pragma unroll
    for (int i = 0; i < 4; i++)
#pragma unroll
      for (int j = 0; j < 2; j++)
        acc[i][j] = __builtin_amdgcn_wmma_f32_16x16x32_bf16(
            false, afr[i], false, bfr[j], (short)0, acc[i][j], false, false);
  }

  float sc = 0.0f;
  if (OUT_MODE == 3 || OUT_MODE == 4) sc = *scale;

#pragma unroll
  for (int i = 0; i < 4; i++) {
#pragma unroll
    for (int j = 0; j < 2; j++) {
      const int m_tile = m_wave + i * 16;
      const int n_tile = n_wave + j * 16;
      if (m_tile >= M || n_tile >= N) continue;   // whole-tile guard (dims % 16 == 0)

      if (OUT_MODE == 1 || OUT_MODE == 3) {
        // direct store: lane owns column n, 8 consecutive rows m
        const int n  = n_tile + r;
        const int mb = m_tile + hi * 8;
        float b = (BIAS_MODE == 2) ? bias[n] : 0.0f;
        if (OUT_MODE == 1) {
          v8bf pk;
#pragma unroll
          for (int v = 0; v < 8; v++) {
            float val = acc[i][j][v];
            if (BIAS_MODE == 1) val += bias[mb + v];
            pk[v] = f2bf(val + b);
          }
          *(v8bf*)((bf16_t*)Out + (size_t)n * ldc + mb) = pk;
        } else { // 3
          size_t idx = (size_t)n * ldc + mb;
          v8f xv = *(const v8f*)(xadd + idx);
          v8f res;
#pragma unroll
          for (int v = 0; v < 8; v++) res[v] = sc * (acc[i][j][v] + b) + 2.0f * xv[v];
          *(v8f*)((float*)Out + idx) = res;
        }
      } else {
        // normal orientation: transpose 16x16 tile through wave-private LDS
        __shared__ float xpose[8][256];
        float* slab = &xpose[wave][0];
#pragma unroll
        for (int v = 0; v < 8; v++) {
          float val = acc[i][j][v];
          if (BIAS_MODE == 1) val += bias[m_tile + hi * 8 + v];
          if (BIAS_MODE == 2) val += bias[n_tile + r];
          slab[(hi * 8 + v) * 16 + r] = val;
        }
        __builtin_amdgcn_wave_barrier();   // LDS in-order per wave; stop reordering
        // lane now owns row r, columns [hi*8, hi*8+8)
        float t[8];
#pragma unroll
        for (int e = 0; e < 8; e++) t[e] = slab[r * 16 + hi * 8 + e];
        __builtin_amdgcn_wave_barrier();
        const size_t idx = (size_t)(m_tile + r) * ldc + n_tile + hi * 8;
        if (OUT_MODE == 0) {
          v8bf pk;
#pragma unroll
          for (int e = 0; e < 8; e++) pk[e] = f2bf(t[e]);
          *(v8bf*)((bf16_t*)Out + idx) = pk;
        } else if (OUT_MODE == 2) {
          v8f res;
#pragma unroll
          for (int e = 0; e < 8; e++) res[e] = t[e];
          *(v8f*)((float*)Out + idx) = res;
        } else { // 4: accumulate
          float* op = (float*)Out + idx;
          v8f cur = *(const v8f*)op;
          v8f res;
#pragma unroll
          for (int e = 0; e < 8; e++) res[e] = cur[e] + sc * t[e];
          *(v8f*)op = res;
        }
      }
    }
  }
}

// ---------------------------------------------------------------------------
// Softmax kernels: one block per row, 256 threads. Row is DMA'd global->LDS by
// the Tensor Data Mover (wave 0 issues, s_wait_tensorcnt, barrier), then the
// block reduces from LDS. Falls back to plain loads without the TDM builtin.
// ---------------------------------------------------------------------------
__global__ __launch_bounds__(256) void k_softmax_bf16_inplace(bf16_t* __restrict__ data, int rowlen) {
  __shared__ bf16_t raw[4096];
  __shared__ float buf[4096];
  __shared__ float red[256];
  bf16_t* p = data + (size_t)blockIdx.x * rowlen;
  const int t = threadIdx.x;
#if HAVE_TDM
  if ((t >> 5) == 0) {
    tdm_load_row_to_lds((unsigned)(uintptr_t)&raw[0], p, (unsigned)rowlen, 1u);
    __builtin_amdgcn_s_wait_tensorcnt(0);
  }
  __syncthreads();
  float lmax = -3.4e38f;
  for (int i = t; i < rowlen; i += 256) { float v = bf2f(raw[i]); buf[i] = v; lmax = fmaxf(lmax, v); }
#else
  float lmax = -3.4e38f;
  for (int i = t; i < rowlen; i += 256) { float v = bf2f(p[i]); buf[i] = v; lmax = fmaxf(lmax, v); }
#endif
  red[t] = lmax; __syncthreads();
  for (int s = 128; s > 0; s >>= 1) { if (t < s) red[t] = fmaxf(red[t], red[t + s]); __syncthreads(); }
  float rmax = red[0]; __syncthreads();
  float lsum = 0.0f;
  for (int i = t; i < rowlen; i += 256) { float e = __expf(buf[i] - rmax); buf[i] = e; lsum += e; }
  red[t] = lsum; __syncthreads();
  for (int s = 128; s > 0; s >>= 1) { if (t < s) red[t] += red[t + s]; __syncthreads(); }
  float inv = 1.0f / red[0];
  for (int i = t; i < rowlen; i += 256) p[i] = f2bf(buf[i] * inv);
}

__global__ __launch_bounds__(256) void k_softmax_f32_to_bf16(const float* __restrict__ in,
                                                             bf16_t* __restrict__ out, int rowlen) {
  __shared__ float buf[4096];
  __shared__ float red[256];
  const float* p = in + (size_t)blockIdx.x * rowlen;
  bf16_t* o = out + (size_t)blockIdx.x * rowlen;
  const int t = threadIdx.x;
#if HAVE_TDM
  if ((t >> 5) == 0) {
    tdm_load_row_to_lds((unsigned)(uintptr_t)&buf[0], p, (unsigned)rowlen, 2u);
    __builtin_amdgcn_s_wait_tensorcnt(0);
  }
  __syncthreads();
  float lmax = -3.4e38f;
  for (int i = t; i < rowlen; i += 256) lmax = fmaxf(lmax, buf[i]);
#else
  float lmax = -3.4e38f;
  for (int i = t; i < rowlen; i += 256) { float v = p[i]; buf[i] = v; lmax = fmaxf(lmax, v); }
#endif
  red[t] = lmax; __syncthreads();
  for (int s = 128; s > 0; s >>= 1) { if (t < s) red[t] = fmaxf(red[t], red[t + s]); __syncthreads(); }
  float rmax = red[0]; __syncthreads();
  float lsum = 0.0f;
  for (int i = t; i < rowlen; i += 256) { float e = __expf(buf[i] - rmax); buf[i] = e; lsum += e; }
  red[t] = lsum; __syncthreads();
  for (int s = 128; s > 0; s >>= 1) { if (t < s) red[t] += red[t + s]; __syncthreads(); }
  float inv = 1.0f / red[0];
  for (int i = t; i < rowlen; i += 256) o[i] = f2bf(buf[i] * inv);
}

// ---------------------------------------------------------------------------
// Host side
// ---------------------------------------------------------------------------
template<int OM, int BM>
static void launch_gemm(const bf16_t* A, const bf16_t* Bt, void* Out,
                        const float* bias, const float* scale, const float* xadd,
                        int M, int N, int K, int lda, int ldbt, int ldc, hipStream_t s) {
  dim3 grid((N + 127) / 128, (M + 127) / 128);
  gemm_bf16_wmma<OM, BM><<<grid, 256, 0, s>>>(A, Bt, Out, bias, scale, xadd,
                                              M, N, K, lda, ldbt, ldc);
}

static void launch_cvt(const float* in, bf16_t* out, int n, hipStream_t s) {
  int blocks = (n + 255) / 256; if (blocks > 2048) blocks = 2048;
  k_f32_to_bf16<<<blocks, 256, 0, s>>>(in, out, n);
}

extern "C" void kernel_launch(void* const* d_in, const int* in_sizes, int n_in,
                              void* d_out, int out_size, void* d_ws, size_t ws_size,
                              hipStream_t stream) {
  (void)in_sizes; (void)n_in; (void)out_size; (void)ws_size;

  const float* x     = (const float*)d_in[0];
  const float* wq    = (const float*)d_in[1];
  const float* bq    = (const float*)d_in[2];
  const float* wk    = (const float*)d_in[3];
  const float* bk    = (const float*)d_in[4];
  const float* wv    = (const float*)d_in[5];
  const float* bv    = (const float*)d_in[6];
  const float* w_so  = (const float*)d_in[7];
  const float* b_so  = (const float*)d_in[8];
  const float* wcq   = (const float*)d_in[9];
  const float* bcq   = (const float*)d_in[10];
  const float* wck   = (const float*)d_in[11];
  const float* bck   = (const float*)d_in[12];
  const float* wcv   = (const float*)d_in[13];
  const float* bcv   = (const float*)d_in[14];
  const float* gamma = (const float*)d_in[15];
  const float* beta  = (const float*)d_in[16];
  float* out = (float*)d_out;

  const int Bq = BQ, C = CCH, IC = ICH, N = NSP;

  // ---- workspace carve (~112 MB) ----
  char* ws = (char*)d_ws;
  size_t off = 0;
  auto carve = [&](size_t bytes) -> void* {
    void* p = (void*)(ws + off);
    off = (off + bytes + 255) & ~(size_t)255;
    return p;
  };
  bf16_t* xT     = (bf16_t*)carve((size_t)Bq * N * C * 2);   // x^T  [B][N][C]
  bf16_t* wq_b   = (bf16_t*)carve((size_t)IC * C * 2);
  bf16_t* wk_b   = (bf16_t*)carve((size_t)IC * C * 2);
  bf16_t* wv_b   = (bf16_t*)carve((size_t)IC * C * 2);
  bf16_t* wcq_b  = (bf16_t*)carve((size_t)C * C * 2);
  bf16_t* wck_b  = (bf16_t*)carve((size_t)C * C * 2);
  bf16_t* wcv_b  = (bf16_t*)carve((size_t)C * C * 2);
  bf16_t* wso_b  = (bf16_t*)carve((size_t)C * IC * 2);       // [C][IC] (already the Bt we need)
  bf16_t* qT     = (bf16_t*)carve((size_t)Bq * N * IC * 2);  // [B][N][IC]
  bf16_t* kT     = (bf16_t*)carve((size_t)Bq * N * IC * 2);  // [B][N][IC]
  bf16_t* vv     = (bf16_t*)carve((size_t)Bq * IC * N * 2);  // [B][IC][N]
  bf16_t* qc     = (bf16_t*)carve((size_t)Bq * C * N * 2);   // [B][C][N]
  bf16_t* kc     = (bf16_t*)carve((size_t)Bq * C * N * 2);   // [B][C][N]
  bf16_t* vcT    = (bf16_t*)carve((size_t)Bq * N * C * 2);   // [B][N][C]
  bf16_t* out_sT = (bf16_t*)carve((size_t)Bq * N * IC * 2);  // [B][N][IC]
  bf16_t* attn_s = (bf16_t*)carve((size_t)N * N * 2);        // reused per batch
  float*  en_c   = (float*) carve((size_t)C * C * 4);        // reused per batch
  bf16_t* attn_c = (bf16_t*)carve((size_t)C * C * 2);        // reused per batch

  // ---- stage 0: precision conversion ----
  {
    dim3 tg(32, 8);
    dim3 gr((N + 31) / 32, (C + 31) / 32);
    for (int b = 0; b < Bq; b++)
      k_transpose_f32_to_bf16<<<gr, tg, 0, stream>>>(
          x + (size_t)b * C * N, xT + (size_t)b * N * C, C, N);  // [C][N] -> [N][C]
  }
  launch_cvt(wq,   wq_b,  IC * C, stream);
  launch_cvt(wk,   wk_b,  IC * C, stream);
  launch_cvt(wv,   wv_b,  IC * C, stream);
  launch_cvt(wcq,  wcq_b, C * C,  stream);
  launch_cvt(wck,  wck_b, C * C,  stream);
  launch_cvt(wcv,  wcv_b, C * C,  stream);
  launch_cvt(w_so, wso_b, C * IC, stream);

  // ---- stage 1: projections (per batch); A = W [Cout,C], Bt = xT [N,C] ----
  for (int b = 0; b < Bq; b++) {
    const bf16_t* xTb = xT + (size_t)b * N * C;
    launch_gemm<1, 1>(wq_b, xTb, qT + (size_t)b * N * IC, bq, nullptr, nullptr,
                      IC, N, C, C, C, IC, stream);               // qT  [N][IC]
    launch_gemm<1, 1>(wk_b, xTb, kT + (size_t)b * N * IC, bk, nullptr, nullptr,
                      IC, N, C, C, C, IC, stream);               // kT  [N][IC]
    launch_gemm<0, 1>(wv_b, xTb, vv + (size_t)b * IC * N, bv, nullptr, nullptr,
                      IC, N, C, C, C, N, stream);                // v   [IC][N]
    launch_gemm<0, 1>(wcq_b, xTb, qc + (size_t)b * C * N, bcq, nullptr, nullptr,
                      C, N, C, C, C, N, stream);                 // qc  [C][N]
    launch_gemm<0, 1>(wck_b, xTb, kc + (size_t)b * C * N, bck, nullptr, nullptr,
                      C, N, C, C, C, N, stream);                 // kc  [C][N]
    launch_gemm<1, 1>(wcv_b, xTb, vcT + (size_t)b * N * C, bcv, nullptr, nullptr,
                      C, N, C, C, C, C, stream);                 // vcT [N][C]
  }

  // ---- stage 2: attention + output (per batch, stream-ordered buffer reuse) ----
  for (int b = 0; b < Bq; b++) {
    const float* xbf = x + (size_t)b * C * N;
    float* outb = out + (size_t)b * C * N;

    // spatial: energy[N,N] = qT @ k   (Bt = kT)
    launch_gemm<0, 0>(qT + (size_t)b * N * IC, kT + (size_t)b * N * IC, attn_s,
                      nullptr, nullptr, nullptr, N, N, IC, IC, IC, N, stream);
    k_softmax_bf16_inplace<<<N, 256, 0, stream>>>(attn_s, N);
    // out_sT[N,IC] = attn @ vT        (Bt = v [IC][N])
    launch_gemm<0, 0>(attn_s, vv + (size_t)b * IC * N, out_sT + (size_t)b * N * IC,
                      nullptr, nullptr, nullptr, N, IC, N, N, N, IC, stream);
    // fused spatial out: out[c][n] = gamma*(out_sT @ w_so^T + b_so)[n][c] + 2*x[c][n]
    launch_gemm<3, 2>(out_sT + (size_t)b * N * IC, wso_b, outb,
                      b_so, gamma, xbf, N, C, IC, IC, IC, N, stream);

    // channel: energy_c[C,C] = qc @ kc^T   (Bt = kc [C][N])
    launch_gemm<2, 0>(qc + (size_t)b * C * N, kc + (size_t)b * C * N, en_c,
                      nullptr, nullptr, nullptr, C, C, N, N, N, C, stream);
    k_softmax_f32_to_bf16<<<C, 256, 0, stream>>>(en_c, attn_c, C);
    // out[c][n] += beta * (attn_c @ vc)[c][n]   (Bt = vcT [N][C])
    launch_gemm<4, 0>(attn_c, vcT + (size_t)b * N * C, outb,
                      nullptr, beta, nullptr, C, N, C, C, C, N, stream);
  }
}
